// SwinTransformerBlock_18356690224064
// MI455X (gfx1250) — matrix operand order repaired and checked by reading, not compiled, not run on hardware
//
#include <hip/hip_runtime.h>
#include <hip/hip_bf16.h>
#include <cmath>

typedef _Float16 h16;
typedef __attribute__((ext_vector_type(8)))  _Float16 v8h;
typedef __attribute__((ext_vector_type(16))) _Float16 v16h;
typedef __attribute__((ext_vector_type(8)))  float    v8f;
typedef __attribute__((ext_vector_type(4)))  int      v4i;

#define WMMA_F16(a,b,c) __builtin_amdgcn_wmma_f32_16x16x32_f16(false,(a),false,(b),(short)0,(c),false,false)

// gfx1250 async global->LDS path (ASYNCcnt); guarded so the file always compiles
#if __has_builtin(__builtin_amdgcn_global_load_async_to_lds_b128) && \
    __has_builtin(__builtin_amdgcn_s_wait_asynccnt)
#define HAVE_ASYNC 1
#else
#define HAVE_ASYNC 0
#endif
#define AS1 __attribute__((address_space(1)))
#define AS3 __attribute__((address_space(3)))

static __device__ __forceinline__ void async_b128(const h16* g, h16* l) {
#if HAVE_ASYNC
  __builtin_amdgcn_global_load_async_to_lds_b128(
      (AS1 v4i*)(AS1 void*)g, (AS3 v4i*)(AS3 void*)l, 0, 0);
#else
  *(v8h*)l = *(const v8h*)g;
#endif
}
static __device__ __forceinline__ void wait_async_keep3() {
#if HAVE_ASYNC
  __builtin_amdgcn_s_wait_asynccnt(3);   // previous tile (3 ops/wave) retired
#endif
}
static __device__ __forceinline__ void wait_async_all() {
#if HAVE_ASYNC
  __builtin_amdgcn_s_wait_asynccnt(0);
#endif
}

static __device__ __forceinline__ v16h cat16(v8h lo, v8h hi) {
  union { v16h v; v8h h[2]; } u; u.h[0] = lo; u.h[1] = hi; return u.v;
}

// ---------------------------------------------------------------------------
// Problem constants
// ---------------------------------------------------------------------------
#define Bb 64
#define Hh 56
#define Ww 56
#define Cc 192
#define HEADS 6
#define HD 32
#define WS 7
#define SHIFT 3
#define NW 64               // windows per image (8x8)
#define TOK 49              // tokens per window
#define NWIN (Bb*NW)        // 4096 windows total
#define MTOT (NWIN*TOK)     // 200704 rows
#define HIDDEN (4*Cc)       // 768

// ---------------------------------------------------------------------------
// fp32 -> fp16 weight conversion
// ---------------------------------------------------------------------------
__global__ void cvt_f32_f16(const float* __restrict__ s, h16* __restrict__ d, int n) {
  int i = blockIdx.x * 256 + threadIdx.x;
  if (i < n) d[i] = (h16)s[i];
}

// ---------------------------------------------------------------------------
// LN1 fused with roll(-SHIFT,-SHIFT) + window partition. One wave per token.
// ---------------------------------------------------------------------------
__global__ __launch_bounds__(256)
void ln1_window(const float* __restrict__ x, const float* __restrict__ g,
                const float* __restrict__ b, h16* __restrict__ xw) {
  const int lane = threadIdx.x & 31, wave = threadIdx.x >> 5;
  const int u   = blockIdx.x * 8 + wave;          // 0..MTOT-1 (grid exact)
  const int win = u / TOK, tok = u % TOK;
  const int bb = win >> 6, wloc = win & 63, wy = wloc >> 3, wx = wloc & 7;
  const int ty = tok / WS, tx = tok % WS;
  const int hh = (wy * WS + ty + SHIFT) % Hh;
  const int ww = (wx * WS + tx + SHIFT) % Ww;
  const float* row = x + ((size_t)bb * (Hh * Ww) + hh * Ww + ww) * Cc;
  float v[6], s = 0.f, s2 = 0.f;
  #pragma unroll
  for (int i = 0; i < 6; i++) { v[i] = row[lane + 32 * i]; s += v[i]; s2 += v[i] * v[i]; }
  #pragma unroll
  for (int o = 16; o >= 1; o >>= 1) { s += __shfl_xor(s, o, 32); s2 += __shfl_xor(s2, o, 32); }
  const float mean = s * (1.f / Cc);
  const float rstd = rsqrtf(s2 * (1.f / Cc) - mean * mean + 1e-5f);
  h16* orow = xw + (size_t)u * Cc;
  #pragma unroll
  for (int i = 0; i < 6; i++) {
    int c = lane + 32 * i;
    orow[c] = (h16)((v[i] - mean) * rstd * g[c] + b[c]);
  }
}

// plain LN2 over x1 rows -> f16
__global__ __launch_bounds__(256)
void ln2_rows(const float* __restrict__ x1, const float* __restrict__ g,
              const float* __restrict__ b, h16* __restrict__ out) {
  const int lane = threadIdx.x & 31, wave = threadIdx.x >> 5;
  const size_t u = (size_t)blockIdx.x * 8 + wave;
  const float* row = x1 + u * Cc;
  float v[6], s = 0.f, s2 = 0.f;
  #pragma unroll
  for (int i = 0; i < 6; i++) { v[i] = row[lane + 32 * i]; s += v[i]; s2 += v[i] * v[i]; }
  #pragma unroll
  for (int o = 16; o >= 1; o >>= 1) { s += __shfl_xor(s, o, 32); s2 += __shfl_xor(s2, o, 32); }
  const float mean = s * (1.f / Cc);
  const float rstd = rsqrtf(s2 * (1.f / Cc) - mean * mean + 1e-5f);
  h16* orow = out + u * Cc;
  #pragma unroll
  for (int i = 0; i < 6; i++) {
    int c = lane + 32 * i;
    orow[c] = (h16)((v[i] - mean) * rstd * g[c] + b[c]);
  }
}

// ---------------------------------------------------------------------------
// Generic f16 GEMM: D[m,n] = sum_k A[m,k] * W[n,k], epilogue fused.
// Block tile 128x64, wave tile 32x32, K step 32, 8 wave32.
// Double-buffered LDS fed by async global->LDS (ASYNCcnt pipeline).
// ---------------------------------------------------------------------------
#define LDA 40   // halves (80B rows: 16B-aligned, conflict-free b128 frag reads)
#define LDB 40

template <class Epi>
__global__ __launch_bounds__(256)
void gemm_f16(const h16* __restrict__ A, const h16* __restrict__ W,
              int M, int N, int K, Epi epi) {
  __shared__ h16 As[2][128 * LDA];
  __shared__ h16 Bs[2][64 * LDB];
  const int t = threadIdx.x;
  const int lane = t & 31, wave = t >> 5;
  const int wm = wave >> 1, wn = wave & 1;
  const int m0 = blockIdx.y * 128, n0 = blockIdx.x * 64;

  // each wave issues exactly 3 async ops per tile: 2 A-chunks + 1 B-chunk
  const int ra0 = t >> 2,        sa0 = t & 3;          // A chunk #t
  const int ra1 = (t + 256) >> 2, sa1 = t & 3;         // A chunk #t+256
  auto stage = [&](int k0, int bufi) {
    async_b128(&A[(size_t)(m0 + ra0) * K + k0 + sa0 * 8], &As[bufi][ra0 * LDA + sa0 * 8]);
    async_b128(&A[(size_t)(m0 + ra1) * K + k0 + sa1 * 8], &As[bufi][ra1 * LDA + sa1 * 8]);
    async_b128(&W[(size_t)(n0 + ra0) * K + k0 + sa0 * 8], &Bs[bufi][ra0 * LDB + sa0 * 8]);
#if !HAVE_ASYNC
    if (k0 + 32 < K) __builtin_prefetch(&A[(size_t)(m0 + ra0) * K + k0 + 32 + sa0 * 8], 0, 3);
#endif
  };

  v8f c[2][2] = {};
  const int nk = K >> 5;
  stage(0, 0);
  for (int kt = 0; kt < nk; ++kt) {
    const int cur = kt & 1;
    if (kt + 1 < nk) { stage((kt + 1) << 5, cur ^ 1); wait_async_keep3(); }
    else             { wait_async_all(); }
    __syncthreads();                       // tile kt visible to all waves

    const int s = lane >> 4, rr = lane & 15, kb = s * 16;
    v16h a[2], bfr[2];
    #pragma unroll
    for (int i = 0; i < 2; i++) {          // A frag: lane holds K=8s..8s+7, 16+8s..23+8s
      const h16* p = &As[cur][(wm * 32 + i * 16 + rr) * LDA];
      a[i] = cat16(*(const v8h*)&p[8 * s], *(const v8h*)&p[16 + 8 * s]);
    }
    #pragma unroll
    for (int j = 0; j < 2; j++) {          // B frag: lane holds K=8s..8s+7, 16+8s..23+8s
      const h16* p = &Bs[cur][(wn * 32 + j * 16 + rr) * LDB];
      bfr[j] = cat16(*(const v8h*)&p[8 * s], *(const v8h*)&p[16 + 8 * s]);
    }
    #pragma unroll
    for (int i = 0; i < 2; i++)
      #pragma unroll
      for (int j = 0; j < 2; j++) c[i][j] = WMMA_F16(a[i], bfr[j], c[i][j]);
    __syncthreads();                       // done reading buf `cur` (refilled at kt+2)
  }
  const int mro = (lane >> 4) * 8, nco = lane & 15;
  #pragma unroll
  for (int i = 0; i < 2; i++)
    #pragma unroll
    for (int j = 0; j < 2; j++)
      #pragma unroll
      for (int e = 0; e < 8; e++)
        epi(m0 + wm * 32 + i * 16 + mro + e, n0 + wn * 32 + j * 16 + nco, c[i][j][e]);
}

// ------------------------- GEMM epilogues ----------------------------------
struct EpiQKV {      // scatter to q/k [tok][hd], v [hd][tok]; fold q scale
  h16* qkv; const float* bias; float scale;
  __device__ void operator()(int m, int n, float acc) const {
    float v = acc + bias[n];
    int win = m / TOK, tok = m % TOK;
    int which = n / Cc, r = n % Cc, head = r >> 5, d = r & 31;
    if (which == 0) v *= scale;
    size_t base = ((size_t)(win * 3 + which) * HEADS + head) * (TOK * HD);
    size_t addr = (which == 2) ? base + (size_t)d * TOK + tok
                               : base + (size_t)tok * HD + d;
    qkv[addr] = (h16)v;
  }
};
struct EpiProj {     // window reverse + roll(+3,+3) + residual -> fp32 x1
  float* x1; const float* bias; const float* x0;
  __device__ void operator()(int m, int n, float acc) const {
    int win = m / TOK, tok = m % TOK;
    int bb = win >> 6, wloc = win & 63, wy = wloc >> 3, wx = wloc & 7;
    int ty = tok / WS, tx = tok % WS;
    int hh = (wy * WS + ty + SHIFT) % Hh, ww = (wx * WS + tx + SHIFT) % Ww;
    size_t p = ((size_t)bb * (Hh * Ww) + hh * Ww + ww) * Cc + n;
    x1[p] = acc + bias[n] + x0[p];
  }
};
struct EpiGelu {     // exact GELU -> f16
  h16* o; const float* bias;
  __device__ void operator()(int m, int n, float acc) const {
    float v = acc + bias[n];
    o[(size_t)m * HIDDEN + n] = (h16)(0.5f * v * (1.f + erff(v * 0.70710678118f)));
  }
};
struct EpiOut {      // + bias + residual -> fp32 output
  float* o; const float* bias; const float* x1;
  __device__ void operator()(int m, int n, float acc) const {
    size_t p = (size_t)m * Cc + n;
    o[p] = acc + bias[n] + x1[p];
  }
};

// ---------------------------------------------------------------------------
// Windowed attention: one block per (window, head). 128 threads = 4 wave32.
// S = Q K^T (64x64 padded, K=32) ; softmax(+rpb+mask) ; O = P V (64x32, K=64)
// ---------------------------------------------------------------------------
#define LQ 40
#define LV 72
#define LP 72
#define LS 68
__global__ __launch_bounds__(128)
void attn_win(const h16* __restrict__ qkv, const float* __restrict__ rpb,
              const int* __restrict__ rel_idx, const float* __restrict__ amask,
              h16* __restrict__ outp) {
  __shared__ h16 Qs[64 * LQ];
  __shared__ h16 Ks[64 * LQ];
  __shared__ h16 Vt[32 * LV];
  __shared__ h16 Pf[64 * LP];
  __shared__ float Sf[64 * LS];
  const int t = threadIdx.x, lane = t & 31, wave = t >> 5;
  const int win = blockIdx.x, head = blockIdx.y;

  for (int i = t; i < 64 * LQ; i += 128) { Qs[i] = (h16)0; Ks[i] = (h16)0; }
  for (int i = t; i < 32 * LV; i += 128) Vt[i] = (h16)0;
  __syncthreads();

  const size_t bq = ((size_t)(win * 3 + 0) * HEADS + head) * (TOK * HD);
  const size_t bk = ((size_t)(win * 3 + 1) * HEADS + head) * (TOK * HD);
  const size_t bv = ((size_t)(win * 3 + 2) * HEADS + head) * (TOK * HD);
  for (int c8 = t; c8 < TOK * 4; c8 += 128) {      // q,k rows: 4 b128 chunks each
    int tk = c8 >> 2, sgm = c8 & 3;
    async_b128(&qkv[bq + tk * HD + sgm * 8], &Qs[tk * LQ + sgm * 8]);
    async_b128(&qkv[bk + tk * HD + sgm * 8], &Ks[tk * LQ + sgm * 8]);
  }
  for (int i = t; i < HD * TOK; i += 128) {        // v already [hd][tok]
    int d = i / TOK, tk = i % TOK;
    Vt[d * LV + tk] = qkv[bv + i];
  }
  wait_async_all();
  __syncthreads();

  { // S = Q K^T : wave owns m-tile `wave`, loops 4 n-tiles, K=32 -> 1 WMMA each
    const int s = lane >> 4, rr = lane & 15, kb = s * 16;
    const h16* pq = &Qs[(wave * 16 + rr) * LQ];
    v16h aq = cat16(*(const v8h*)&pq[8 * s], *(const v8h*)&pq[16 + 8 * s]);
    #pragma unroll
    for (int nt = 0; nt < 4; nt++) {
      const h16* pk = &Ks[(nt * 16 + rr) * LQ];
      v16h bk2 = cat16(*(const v8h*)&pk[8 * s], *(const v8h*)&pk[16 + 8 * s]);
      v8f acc = {};
      acc = WMMA_F16(aq, bk2, acc);
      #pragma unroll
      for (int e = 0; e < 8; e++)
        Sf[(wave * 16 + e + 8 * s) * LS + nt * 16 + rr] = acc[e];
    }
  }
  __syncthreads();

  if (t < 64) {                                     // softmax per row (in LDS)
    const int r = t;
    if (r < TOK) {
      const float* mrow = amask + ((size_t)(win & 63) * TOK + r) * TOK;
      float mx = -1e30f;
      for (int cc = 0; cc < TOK; cc++) {
        float bias = rpb[rel_idx[r * TOK + cc] * HEADS + head];
        float vv = Sf[r * LS + cc] + bias + mrow[cc];
        Sf[r * LS + cc] = vv;
        mx = fmaxf(mx, vv);
      }
      float ssum = 0.f;
      for (int cc = 0; cc < TOK; cc++) { float e = __expf(Sf[r * LS + cc] - mx); Sf[r * LS + cc] = e; ssum += e; }
      float inv = 1.f / ssum;
      for (int cc = 0; cc < TOK; cc++) Pf[r * LP + cc] = (h16)(Sf[r * LS + cc] * inv);
      for (int cc = TOK; cc < 64; cc++) Pf[r * LP + cc] = (h16)0;
    } else {
      for (int cc = 0; cc < 64; cc++) Pf[r * LP + cc] = (h16)0;
    }
  }
  __syncthreads();

  { // O = P V : wave owns m-tile `wave`, 2 n-tiles, K=64 -> 2 WMMAs each
    const int s = lane >> 4, rr = lane & 15;
    #pragma unroll
    for (int j = 0; j < 2; j++) {
      v8f acc = {};
      #pragma unroll
      for (int kk = 0; kk < 64; kk += 32) {
        const h16* pp = &Pf[(wave * 16 + rr) * LP + kk];
        v16h ap = cat16(*(const v8h*)&pp[8 * s], *(const v8h*)&pp[16 + 8 * s]);
        const h16* pv = &Vt[(j * 16 + rr) * LV + kk];
        v16h bv2 = cat16(*(const v8h*)&pv[8 * s], *(const v8h*)&pv[16 + 8 * s]);
        acc = WMMA_F16(ap, bv2, acc);
      }
      #pragma unroll
      for (int e = 0; e < 8; e++) {
        int tq = wave * 16 + e + 8 * s;
        if (tq < TOK)
          outp[((size_t)win * TOK + tq) * Cc + head * HD + j * 16 + rr] = (h16)acc[e];
      }
    }
  }
}

// ---------------------------------------------------------------------------
extern "C" void kernel_launch(void* const* d_in, const int* in_sizes, int n_in,
                              void* d_out, int out_size, void* d_ws, size_t ws_size,
                              hipStream_t stream) {
  const float* x       = (const float*)d_in[0];
  const float* n1g     = (const float*)d_in[1];
  const float* n1b     = (const float*)d_in[2];
  const float* qkv_w   = (const float*)d_in[3];
  const float* qkv_b   = (const float*)d_in[4];
  const float* rpb     = (const float*)d_in[5];
  const float* proj_w  = (const float*)d_in[6];
  const float* proj_b  = (const float*)d_in[7];
  const float* n2g     = (const float*)d_in[8];
  const float* n2b     = (const float*)d_in[9];
  const float* fc1_w   = (const float*)d_in[10];
  const float* fc1_b   = (const float*)d_in[11];
  const float* fc2_w   = (const float*)d_in[12];
  const float* fc2_b   = (const float*)d_in[13];
  const int*   rel_idx = (const int*)  d_in[14];
  const float* amask   = (const float*)d_in[15];
  float* out = (float*)d_out;

  // workspace partition (regions reused across phases)
  char* ws = (char*)d_ws;
  size_t off = 0;
  auto carve = [&](size_t bytes) -> void* {
    off = (off + 255) & ~(size_t)255;
    void* p = ws + off; off += bytes; return p;
  };
  float* x1   = (float*)carve((size_t)MTOT * Cc * sizeof(float));        // residual mid
  h16* bufA   = (h16*)  carve((size_t)MTOT * HIDDEN * sizeof(h16));      // xw -> attn_out -> mlp mid
  h16* bufB   = (h16*)  carve((size_t)MTOT * 3 * Cc * sizeof(h16));      // qkv -> ln2 out
  h16* wqkv   = (h16*)  carve((size_t)3 * Cc * Cc * sizeof(h16));
  h16* wproj  = (h16*)  carve((size_t)Cc * Cc * sizeof(h16));
  h16* wfc1   = (h16*)  carve((size_t)HIDDEN * Cc * sizeof(h16));
  h16* wfc2   = (h16*)  carve((size_t)Cc * HIDDEN * sizeof(h16));
  h16* xw       = bufA;
  h16* attn_out = bufA;
  h16* mlp_mid  = bufA;
  h16* qkvbuf   = bufB;
  h16* ln2_out  = bufB;

  // weight conversion fp32 -> fp16
  auto cvt = [&](const float* s, h16* d, int n) {
    cvt_f32_f16<<<(n + 255) / 256, 256, 0, stream>>>(s, d, n);
  };
  cvt(qkv_w, wqkv, 3 * Cc * Cc);
  cvt(proj_w, wproj, Cc * Cc);
  cvt(fc1_w, wfc1, HIDDEN * Cc);
  cvt(fc2_w, wfc2, Cc * HIDDEN);

  const dim3 blk256(256), blk128(128);
  // 1) LN1 + shift + window partition
  ln1_window<<<MTOT / 8, blk256, 0, stream>>>(x, n1g, n1b, xw);
  // 2) QKV GEMM (200704 x 576 x 192)
  gemm_f16<<<dim3(576 / 64, MTOT / 128), blk256, 0, stream>>>(
      xw, wqkv, MTOT, 3 * Cc, Cc, EpiQKV{qkvbuf, qkv_b, 0.17677669529663687f});
  // 3) windowed attention
  attn_win<<<dim3(NWIN, HEADS), blk128, 0, stream>>>(qkvbuf, rpb, rel_idx, amask, attn_out);
  // 4) proj GEMM + reverse-shift + residual
  gemm_f16<<<dim3(Cc / 64, MTOT / 128), blk256, 0, stream>>>(
      attn_out, wproj, MTOT, Cc, Cc, EpiProj{x1, proj_b, x});
  // 5) LN2
  ln2_rows<<<MTOT / 8, blk256, 0, stream>>>(x1, n2g, n2b, ln2_out);
  // 6) fc1 + GELU (200704 x 768 x 192)
  gemm_f16<<<dim3(HIDDEN / 64, MTOT / 128), blk256, 0, stream>>>(
      ln2_out, wfc1, MTOT, HIDDEN, Cc, EpiGelu{mlp_mid, fc1_b});
  // 7) fc2 + residual (200704 x 192 x 768)
  gemm_f16<<<dim3(Cc / 64, MTOT / 128), blk256, 0, stream>>>(
      mlp_mid, wfc2, MTOT, Cc, HIDDEN, EpiOut{out, fc2_b, x1});
}
